// WindowAttention_34437047779507
// MI455X (gfx1250) — matrix operand order repaired and checked by reading, not compiled, not run on hardware
//
#include <hip/hip_runtime.h>
#include <math.h>

// ---------------------------------------------------------------------------
// Fused window attention for MI455X (gfx1250), wave32 + WMMA f16->f32.
// One workgroup (256 threads = 8 wave32) per window; all intermediates in LDS.
// Round 3: usew16 hoisted out of the K-loop via template (clean hot path),
// nontemporal final stores, async global->LDS DMA staging of x kept.
// ---------------------------------------------------------------------------

typedef __attribute__((ext_vector_type(16))) _Float16 v16h;
typedef __attribute__((ext_vector_type(8)))  float    v8f;

#define DIMC   180
#define HEADS  6
#define HDIM   30
#define NTOK   49
#define NWIN   1024
#define NBLK   4096
#define KP     192          // padded DIM (multiple of 32)
#define MP     64           // padded token rows (multiple of 16)
#define W_GAIN 0.074535599249993f   // 1/sqrt(180)
#define ASCALE 0.182574185835055f   // 1/sqrt(30)

union AF { v16h v; _Float16 h[16]; int4 q2[2]; };

static __device__ __forceinline__ v8f wmma16(v16h a, v16h b, v8f c) {
  // D = A(16x32 f16) * B(32x16 f16) + C(16x16 f32)
  return __builtin_amdgcn_wmma_f32_16x16x32_f16(false, a, false, b,
                                                (short)0, c, false, false);
}

// A operand (16x32, f16) from a row-major matrix [row][K], stride ld (halves).
// Lanes 0-15 -> K k0..k0+15, lanes 16-31 -> K k0+16..k0+31; M = lane & 15.
static __device__ __forceinline__ v16h load_a_lds(const _Float16* base, int ld,
                                                  int m0, int k0, int lane) {
  const int g = (lane >> 4) & 1, r = lane & 15;
  const _Float16* p = base + (m0 + r) * ld + k0 + g * 16;
  AF f;
  f.q2[0] = *reinterpret_cast<const int4*>(p);        // 8 halves
  f.q2[1] = *reinterpret_cast<const int4*>(p + 8);    // 8 halves
  return f.v;
}

// B operand (32x16, f16) where the source is stored [col][K] (K contiguous).
// Lanes 0-15 hold K=k0..k0+15, lanes 16-31 K=k0+16..k0+31; N = lane & 15.
// Works for LDS (kh) and for the global f16 weight workspace.
static __device__ __forceinline__ v16h load_b_cm(const _Float16* base, int ld,
                                                 int k0, int n0, int lane) {
  const int g = (lane >> 4) & 1, r = lane & 15;
  const _Float16* p = base + (n0 + r) * ld + k0 + 16 * g;
  AF f;
  f.q2[0] = *reinterpret_cast<const int4*>(p);
  f.q2[1] = *reinterpret_cast<const int4*>(p + 8);
  return f.v;
}

// B operand (32x16, f16) where the source is stored [K][col] (strided gather).
static __device__ __forceinline__ v16h load_b_rm(const _Float16* base, int ld,
                                                 int k0, int n0, int lane) {
  const int g = (lane >> 4) & 1, r = lane & 15;
  const _Float16* p = base + (k0 + 16 * g) * ld + n0 + r;
  AF f;
#pragma unroll
  for (int i = 0; i < 16; ++i) f.h[i] = p[i * ld];
  return f.v;
}

// Fallback B operand from a global fp32 weight (w[o][k], row-contiguous in k):
// B[k][o] = w[o][k] * gain, zero outside the real 180x180 extent.
static __device__ __forceinline__ v16h load_b_wf32(const float* __restrict__ w,
                                                   int k0, int o0, int lane) {
  const int g = (lane >> 4) & 1, r = lane & 15;
  const int o = o0 + r, kb = k0 + 16 * g;
  AF f;
  if (o < DIMC) {
    const float* p = w + o * DIMC + kb;
#pragma unroll
    for (int i = 0; i < 16; ++i)
      f.h[i] = (kb + i < DIMC) ? (_Float16)(p[i] * W_GAIN) : (_Float16)0.0f;
  } else {
#pragma unroll
    for (int i = 0; i < 16; ++i) f.h[i] = (_Float16)0.0f;
  }
  return f.v;
}

// 64x16 output strip: A (64x192 in LDS) x B-column-strip of one weight.
// USE16: B fragments from the pre-converted f16 workspace (2x b128 loads,
// zero VALU). Fallback converts fp32 weights in-flight.
template <bool USE16>
static __device__ __forceinline__ void gemm_strip_192(
    const _Float16* __restrict__ A, const _Float16* __restrict__ W16,
    const float* __restrict__ Wf32, int ot, int lane,
    v8f& c0, v8f& c1, v8f& c2, v8f& c3) {
  v16h bf = USE16 ? load_b_cm(W16, KP, 0, ot * 16, lane)
                  : load_b_wf32(Wf32, 0, ot * 16, lane);
#pragma unroll
  for (int kk = 0; kk < KP; kk += 32) {
    v16h bnext = bf;
    if (kk + 32 < KP)
      bnext = USE16 ? load_b_cm(W16, KP, kk + 32, ot * 16, lane)
                    : load_b_wf32(Wf32, kk + 32, ot * 16, lane);
    c0 = wmma16(load_a_lds(A, KP, 0,  kk, lane), bf, c0);
    c1 = wmma16(load_a_lds(A, KP, 16, kk, lane), bf, c1);
    c2 = wmma16(load_a_lds(A, KP, 32, kk, lane), bf, c2);
    c3 = wmma16(load_a_lds(A, KP, 48, kk, lane), bf, c3);
    bf = bnext;
  }
}

static __device__ __forceinline__ void store_head_tile(_Float16* dst, v8f c,
                                                       int mbase, int h, int d,
                                                       float bv, int g) {
#pragma unroll
  for (int j = 0; j < 8; ++j)
    dst[(h * MP + mbase + g * 8 + j) * 32 + d] = (_Float16)(c[j] + bv);
}

// ---------------------------------------------------------------------------
// Prep kernel: qw/kw/vw/pw (fp32, 180x180) -> gain-scaled, zero-padded f16
// workspace [4][192][192], layout [o][k] so B fragments are two b128 loads.
// ---------------------------------------------------------------------------
__global__ __launch_bounds__(256) void wprep_kernel(
    const float* __restrict__ qw, const float* __restrict__ kw,
    const float* __restrict__ vw, const float* __restrict__ pw,
    _Float16* __restrict__ wsf16) {
  const int idx = blockIdx.x * 256 + threadIdx.x;
  if (idx >= 4 * KP * KP) return;
  const int w = idx / (KP * KP);
  const int rem = idx % (KP * KP);
  const int o = rem / KP, k = rem % KP;
  const float* src = (w == 0) ? qw : (w == 1) ? kw : (w == 2) ? vw : pw;
  const float v = (o < DIMC && k < DIMC) ? src[o * DIMC + k] * W_GAIN : 0.0f;
  wsf16[idx] = (_Float16)v;
}

// ---------------------------------------------------------------------------
// Main fused kernel.
// ---------------------------------------------------------------------------
__global__ __launch_bounds__(256) void winattn_kernel(
    const float* __restrict__ x,    const float* __restrict__ maskp,
    const float* __restrict__ mwin,
    const float* __restrict__ qw,   const float* __restrict__ qb,
    const float* __restrict__ kw,   const float* __restrict__ kb,
    const float* __restrict__ vw,   const float* __restrict__ vb,
    const float* __restrict__ pw,   const float* __restrict__ pb,
    const _Float16* __restrict__ w16, int usew16,
    float* __restrict__ out) {
  extern __shared__ __align__(16) char smem[];
  // f16 zone (pre-zeroed => implicit M/K padding everywhere)
  _Float16* xh  = (_Float16*)smem;          // [64][192] x as f16
  _Float16* nxh = xh  + MP * KP;            // [64][192] normalized x
  _Float16* qh  = nxh + MP * KP;            // [6][64][32] per-head q
  _Float16* kh  = qh  + HEADS * MP * 32;    // [6][64][32] per-head k
  _Float16* vh  = kh  + HEADS * MP * 32;    // [6][64][32] per-head v
  _Float16* oh  = vh  + HEADS * MP * 32;    // [64][192] attn output (proj input)
  _Float16* Pm  = oh  + MP * KP;            // [6][64][64] softmax probs
  // f32 zone (first reused as the async x staging buffer, then raw scores)
  float* S    = (float*)(Pm + HEADS * MP * 64);   // [6][49][64] scores / stage
  float* win  = S + HEADS * NTOK * 64;            // [64] -100/0 key penalty
  float* misc = win + 64;                         // [4]

  const int tid  = threadIdx.x;
  const int lane = tid & 31;          // wave32
  const int wid  = tid >> 5;          // 8 waves
  const int b    = blockIdx.x;
  const bool u16 = (usew16 != 0);

  // ---- phase 0: zero f16 zone; async-DMA x[b] into LDS stage -------------
  {
    uint32_t* z = (uint32_t*)smem;
#pragma unroll 4
    for (int i = tid; i < 49152; i += 256) z[i] = 0u;   // 98304 halves
  }
  {
    // 49*180 floats = 2205 16-byte chunks, copied by async global->LDS DMA
    const float* gx = x + (size_t)b * NTOK * DIMC;
    for (int i = tid; i < (NTOK * DIMC) / 4; i += 256) {
      const unsigned loff = (unsigned)(size_t)(const void*)(S + i * 4);
      const float* gp = gx + i * 4;
      asm volatile("global_load_async_to_lds_b128 %0, %1, off"
                   :: "v"(loff), "v"(gp) : "memory");
    }
  }
  if (tid < NTOK) {
    win[tid] = (mwin[(size_t)b * NTOK + tid] == 0.0f) ? -100.0f : 0.0f;
  } else if (tid < MP) {
    win[tid] = 0.0f;
  }
  if (tid == 0) {
    float s = 0.0f;
    for (int m = 0; m < NTOK; ++m) s += mwin[(size_t)b * NTOK + m];
    misc[0] = fminf(fmaxf(s, 0.0f), 1.0f);
  }
  asm volatile("s_wait_asynccnt 0" ::: "memory");
  __syncthreads();

  // ---- phase 0b: row L2 norms from staged x, emit f16 x / norm_x ---------
  if (tid < NTOK) {
    const float* xr = S + tid * DIMC;   // staged row in LDS
    float ss = 0.0f;
    for (int c = 0; c < DIMC; ++c) {
      const float v = xr[c];
      ss += v * v;
      xh[tid * KP + c] = (_Float16)v;
    }
    const float inv = 1.0f / fmaxf(sqrtf(ss), 1e-12f);
    for (int c = 0; c < DIMC; ++c)
      nxh[tid * KP + c] = (_Float16)(xr[c] * inv);
  }
  __syncthreads();

  // second output: mw_new = clip(sum, 0, 1) tiled over N
  if (tid < NTOK)
    __builtin_nontemporal_store(misc[0],
        out + (size_t)NBLK * NTOK * DIMC + (size_t)b * NTOK + tid);

  // ---- phase 1: q = nx@WqT, k = nx@WkT, v = x@WvT ------------------------
  // Column strips: 3 mats x 12 o-tiles = 36 jobs; each reuses the B fragment
  // across all 4 M-tiles; usew16 branch hoisted to strip granularity.
  for (int job = wid; job < 36; job += 8) {
    const int mat = job / 12;
    const int ot  = job % 12;
    const _Float16* A  = (mat == 2) ? xh : nxh;
    const float* Wf    = (mat == 0) ? qw : (mat == 1) ? kw : vw;
    const float* bias  = (mat == 0) ? qb : (mat == 1) ? kb : vb;
    _Float16* dst      = (mat == 0) ? qh : (mat == 1) ? kh : vh;
    const _Float16* W16 = w16 + (size_t)mat * KP * KP;
    v8f c0 = {}, c1 = {}, c2 = {}, c3 = {};
    if (u16) gemm_strip_192<true >(A, W16, Wf, ot, lane, c0, c1, c2, c3);
    else     gemm_strip_192<false>(A, W16, Wf, ot, lane, c0, c1, c2, c3);
    const int g = (lane >> 4) & 1, r = lane & 15;
    const int o = ot * 16 + r;
    if (o < DIMC) {
      const int h = o / HDIM, d = o % HDIM;   // head-padded store
      const float bv = bias[o];
      store_head_tile(dst, c0, 0,  h, d, bv, g);
      store_head_tile(dst, c1, 16, h, d, bv, g);
      store_head_tile(dst, c2, 32, h, d, bv, g);
      store_head_tile(dst, c3, 48, h, d, bv, g);
    }
  }
  __syncthreads();

  // ---- phase 2a: scores S = q@kT * scale + mask + win (per head) ---------
  // Strips: 6 heads x 4 N-tiles = 24 jobs; B fragment reused over 4 M-tiles.
  for (int job = wid; job < 24; job += 8) {
    const int h  = job / 4;
    const int nt = job % 4;
    const v16h bf = load_b_cm(kh + h * MP * 32, 32, 0, nt * 16, lane);
    const int g = (lane >> 4) & 1, r = lane & 15;
    const int n = nt * 16 + r;
    const float wpen = win[n];
    const float* mrow = maskp + (size_t)(b & (NWIN - 1)) * NTOK * NTOK;
#pragma unroll
    for (int mt = 0; mt < 4; ++mt) {
      v8f c = {};
      c = wmma16(load_a_lds(qh + h * MP * 32, 32, mt * 16, 0, lane), bf, c);
#pragma unroll
      for (int j = 0; j < 8; ++j) {
        const int m = mt * 16 + g * 8 + j;
        if (m < NTOK) {
          float v = c[j] * ASCALE + wpen;
          if (n < NTOK) v += mrow[m * NTOK + n];
          S[(h * NTOK + m) * 64 + n] = v;
        }
      }
    }
  }
  __syncthreads();

  // ---- phase 2b: fp32 softmax over the 49 valid keys ---------------------
  for (int row = tid; row < HEADS * NTOK; row += 256) {
    const int h = row / NTOK, m = row % NTOK;
    const float* sr = S + (h * NTOK + m) * 64;
    float mx = -3.0e38f;
    for (int n = 0; n < NTOK; ++n) mx = fmaxf(mx, sr[n]);
    float sum = 0.0f;
    for (int n = 0; n < NTOK; ++n) sum += expf(sr[n] - mx);
    const float inv = 1.0f / sum;
    _Float16* pr = Pm + (h * MP + m) * 64;
    for (int n = 0; n < NTOK; ++n)
      pr[n] = (_Float16)(expf(sr[n] - mx) * inv);   // cols 49..63 stay 0
  }
  __syncthreads();

  // ---- phase 3: O = P @ v (per head), packed into oh[64][192] ------------
  // Strips: 6 heads x 2 D-tiles = 12 jobs; the two strided v-gathers are
  // loaded once and reused over all 4 M-tiles.
  for (int job = wid; job < 12; job += 8) {
    const int h  = job / 2;
    const int dt = job % 2;
    const v16h b0 = load_b_rm(vh + h * MP * 32, 32, 0,  dt * 16, lane);
    const v16h b1 = load_b_rm(vh + h * MP * 32, 32, 32, dt * 16, lane);
    const int g = (lane >> 4) & 1, r = lane & 15;
    const int d = dt * 16 + r;
#pragma unroll
    for (int mt = 0; mt < 4; ++mt) {
      v8f c = {};
      c = wmma16(load_a_lds(Pm + h * MP * 64, 64, mt * 16, 0,  lane), b0, c);
      c = wmma16(load_a_lds(Pm + h * MP * 64, 64, mt * 16, 32, lane), b1, c);
      if (d < HDIM) {
#pragma unroll
        for (int j = 0; j < 8; ++j) {
          const int m = mt * 16 + g * 8 + j;
          if (m < NTOK) oh[m * KP + h * HDIM + d] = (_Float16)c[j];
        }
      }
    }
  }
  __syncthreads();

  // ---- phase 4: out = O @ WpT * gain + pb --------------------------------
  // Strips: 12 o-tiles; B fragment reused over all 4 M-tiles; NT stores.
  for (int job = wid; job < 12; job += 8) {
    const int ot = job;
    const _Float16* W16 = w16 + (size_t)3 * KP * KP;
    v8f c0 = {}, c1 = {}, c2 = {}, c3 = {};
    if (u16) gemm_strip_192<true >(oh, W16, pw, ot, lane, c0, c1, c2, c3);
    else     gemm_strip_192<false>(oh, W16, pw, ot, lane, c0, c1, c2, c3);
    const int g = (lane >> 4) & 1, r = lane & 15;
    const int o = ot * 16 + r;
    if (o < DIMC) {
      const float bv = pb[o];
      float* orow = out + (size_t)b * NTOK * DIMC;
      const v8f cc[4] = {c0, c1, c2, c3};
#pragma unroll
      for (int mt = 0; mt < 4; ++mt) {
#pragma unroll
        for (int j = 0; j < 8; ++j) {
          const int m = mt * 16 + g * 8 + j;
          if (m < NTOK)
            __builtin_nontemporal_store(cc[mt][j] + bv, orow + m * DIMC + o);
        }
      }
    }
  }
}

extern "C" void kernel_launch(void* const* d_in, const int* in_sizes, int n_in,
                              void* d_out, int out_size, void* d_ws, size_t ws_size,
                              hipStream_t stream) {
  const float* x    = (const float*)d_in[0];
  const float* mask = (const float*)d_in[1];
  const float* mwin = (const float*)d_in[2];
  const float* qw   = (const float*)d_in[3];
  const float* qb   = (const float*)d_in[4];
  const float* kw   = (const float*)d_in[5];
  const float* kb   = (const float*)d_in[6];
  const float* vw   = (const float*)d_in[7];
  const float* vb   = (const float*)d_in[8];
  const float* pw   = (const float*)d_in[9];
  const float* pb   = (const float*)d_in[10];
  float* out = (float*)d_out;

  const size_t wbytes = (size_t)4 * KP * KP * sizeof(_Float16);   // 294912 B
  const int usew16 = (ws_size >= wbytes) ? 1 : 0;
  _Float16* w16 = (_Float16*)d_ws;
  if (usew16) {
    const int total = 4 * KP * KP;
    wprep_kernel<<<(total + 255) / 256, 256, 0, stream>>>(qw, kw, vw, pw, w16);
  }

  // LDS: 98304 f16 (196608 B) + (6*49*64 + 64 + 4) f32 (75536 B) = 272144 B
  const size_t smem_bytes = (size_t)98304 * 2 + (size_t)(HEADS * NTOK * 64 + 64 + 4) * 4;

  winattn_kernel<<<NBLK, 256, smem_bytes, stream>>>(
      x, mask, mwin, qw, qb, kw, kb, vw, vb, pw, pb, w16, usew16, out);
}
